// STDPMNIST_89601607729410
// MI455X (gfx1250) — matrix operand order repaired and checked by reading, not compiled, not run on hardware
//
#include <hip/hip_runtime.h>
#include <hip/hip_bf16.h>

typedef _Float16 v16h __attribute__((ext_vector_type(16)));
typedef _Float16 v8h  __attribute__((ext_vector_type(8)));
typedef float    v8f  __attribute__((ext_vector_type(8)));
typedef unsigned int u32x4 __attribute__((ext_vector_type(4)));
typedef int          i32x8 __attribute__((ext_vector_type(8)));
typedef int          i32x4 __attribute__((ext_vector_type(4)));

#define T_DIM   32
#define HW      256
#define C1_OUT  30
#define POOL    129          // pooled spatial size
#define PADH    131          // 1 + 129 + 1 rows
#define PADW    136          // 131 cols padded to 136 (16B-friendly)
#define CPAD    32           // channels 30 -> 32
#define OH      127          // conv2 output H = W
#define NTILES  7            // 100 output channels padded to 112 = 7*16
#define NPAD    112
#define TAPS    25

#define POOLED_HALVES ((long)T_DIM * PADH * PADW * CPAD)   // 18,243,584 halves = 36.5 MB
#define BPACK_HALVES  (TAPS * NTILES * 32 * 16)            // 89,600 halves = 175 KB
#define ROWBYTES      (PADW * CPAD * 2)                    // 8704 B per pooled row
#define ROWQ          (ROWBYTES / 8)                       // 1088 qwords per row

// ---------------------------------------------------------------------------
// prep: zero pooled buffer (incl. all padding/slack), zero d_out, pack w2 into
// WMMA B-fragment order: bpack[tap][ntile][lane][e] ; N = nt*16 + (lane&15),
// K(=channel) = 16*(lane>>4) + e  (contiguous k order, same as the A loads).
// ---------------------------------------------------------------------------
__global__ void prep_kernel(const float* __restrict__ w2,
                            _Float16* __restrict__ pooled,
                            _Float16* __restrict__ bpack,
                            float* __restrict__ out) {
  long idx = (long)blockIdx.x * blockDim.x + threadIdx.x;
  const long nvec = POOLED_HALVES / 8;        // 16B chunks
  if (idx < nvec) {
    v8h z = {};
    *reinterpret_cast<v8h*>(pooled + idx * 8) = z;
  }
  if (idx < BPACK_HALVES) {
    int i    = (int)idx;
    int tap  = i / (NTILES * 512);
    int r1   = i % (NTILES * 512);
    int nt   = r1 / 512;
    int r2   = r1 % 512;
    int lane = r2 / 16;
    int e    = r2 % 16;
    int n = nt * 16 + (lane & 15);
    int c = ((lane >> 4) << 4) + e;
    int kh = tap / 5, kw = tap % 5;
    float v = 0.0f;
    if (n < 100 && c < 30) v = w2[(((n * 30 + c) * 5 + kh) * 5) + kw];
    bpack[i] = (_Float16)v;
  }
  if (idx < 100) out[idx] = 0.0f;
}

// ---------------------------------------------------------------------------
// conv1 + fire(15) + 2x2/s2/p1 maxpool (windows disjoint; binary max = OR).
// One thread per pooled pixel; writes a 32-channel f16 vector (channel-last).
// ---------------------------------------------------------------------------
__global__ void conv1_fire_pool_kernel(const float* __restrict__ in,
                                       const float* __restrict__ w1,
                                       _Float16* __restrict__ pooled) {
  __shared__ float wT[50 * 32];   // transposed weights [tap][o], broadcast reads
  for (int i = threadIdx.x; i < 50 * 32; i += blockDim.x) {
    int tap = i >> 5, o = i & 31;
    wT[i] = (o < C1_OUT) ? w1[o * 50 + tap] : 0.0f;
  }
  __syncthreads();

  long pid = (long)blockIdx.x * blockDim.x + threadIdx.x;
  const long total = (long)T_DIM * POOL * POOL;
  if (pid >= total) return;
  int t  = (int)(pid / (POOL * POOL));
  int rr = (int)(pid % (POOL * POOL));
  int pi = rr / POOL, pj = rr % POOL;

  unsigned mask = 0u;
  for (int dy = 0; dy < 2; ++dy) {
    int y = 2 * pi - 1 + dy;
    if (y < 0 || y >= HW) continue;
    for (int dx = 0; dx < 2; ++dx) {
      int x = 2 * pj - 1 + dx;
      if (x < 0 || x >= HW) continue;
      float pot[C1_OUT];
#pragma unroll
      for (int o = 0; o < C1_OUT; ++o) pot[o] = 0.0f;
      for (int c = 0; c < 2; ++c) {
        const float* inc = in + (long)(t * 2 + c) * HW * HW;
#pragma unroll
        for (int kh = 0; kh < 5; ++kh) {
          int iy = y + kh - 2;
          if (iy < 0 || iy >= HW) continue;
          const float* row = inc + (long)iy * HW;
#pragma unroll
          for (int kw = 0; kw < 5; ++kw) {
            int ix = x + kw - 2;
            float s = (ix >= 0 && ix < HW) ? row[ix] : 0.0f;
            const float* wrow = &wT[(c * 25 + kh * 5 + kw) << 5];
#pragma unroll
            for (int o = 0; o < C1_OUT; ++o) pot[o] = fmaf(wrow[o], s, pot[o]);
          }
        }
      }
      unsigned m = 0u;
#pragma unroll
      for (int o = 0; o < C1_OUT; ++o) m |= (pot[o] > 15.0f) ? (1u << o) : 0u;
      mask |= m;
    }
  }

  const _Float16 one = (_Float16)1.0f, zer = (_Float16)0.0f;
  v8h v0, v1, v2, v3;
#pragma unroll
  for (int o = 0; o < 8; ++o) {
    v0[o] = ((mask >> (o +  0)) & 1u) ? one : zer;
    v1[o] = ((mask >> (o +  8)) & 1u) ? one : zer;
    v2[o] = ((mask >> (o + 16)) & 1u) ? one : zer;
    v3[o] = ((mask >> (o + 24)) & 1u) ? one : zer;   // bits 30..31 are 0
  }
  long base = (((long)t * PADH + (pi + 1)) * PADW + (pj + 1)) * CPAD;
  v8h* dst = reinterpret_cast<v8h*>(pooled + base);
  dst[0] = v0; dst[1] = v1; dst[2] = v2; dst[3] = v3;
}

// ---------------------------------------------------------------------------
// conv2 as 25 shifted K=32 GEMMs on WMMA + fire(10) + global-any reduction.
// Slab staging via the Tensor Data Mover: one 2-D tile descriptor (1088 qwords
// x 5 rows, stride = row pitch) issued by wave 0, completed with
// s_wait_tensorcnt, then a workgroup barrier releases the compute waves.
// ---------------------------------------------------------------------------
__global__ void conv2_wmma_kernel(const _Float16* __restrict__ pooled,
                                  const _Float16* __restrict__ bpack,
                                  float* __restrict__ out) {
  __shared__ alignas(16) _Float16 slab[5 * PADW * CPAD];  // 43,520 B: 5 input rows
  __shared__ int sflag[NPAD];

  const int t   = blockIdx.x / OH;
  const int oy  = blockIdx.x % OH;
  const int tid = threadIdx.x;

  if (tid < NPAD) sflag[tid] = 0;

  // --- TDM: async-load 5 contiguous channel-last rows into LDS -------------
  const _Float16* src = pooled + ((long)t * PADH + oy) * (long)(PADW * CPAD);
  if (tid < 32) {   // wave 0 issues the tensor DMA (wave-level op, EXEC ignored)
    const unsigned lds_off = (unsigned)(uintptr_t)(void*)slab;   // low 32 = LDS offset
    const unsigned long long ga = (unsigned long long)(uintptr_t)src;
    u32x4 g0;
    g0[0] = 1u;                                  // count=1, user descriptor
    g0[1] = lds_off;                             // lds_addr
    g0[2] = (unsigned)ga;                        // global_addr[31:0]
    g0[3] = (unsigned)((ga >> 32) & 0x01FFFFFFu) // global_addr[56:32]
            | 0x80000000u;                       // type=2 ("image") in bits 127:126
    i32x8 g1;
    g1[0] = (3 << 16);                    // workgroup_mask=0, data_size=8B
    g1[1] = (int)((ROWQ & 0xFFFF) << 16); // tensor_dim0[15:0] (1088 qwords/row)
    g1[2] = (int)(5u << 16);              // tensor_dim0[31:16]=0 | tensor_dim1=5
    g1[3] = (int)(ROWQ << 16);            // tensor_dim1[31:16]=0 | tile_dim0=1088
    g1[4] = 5;                            // tile_dim1=5 rows, tile_dim2=0
    g1[5] = (int)ROWQ;                    // tensor_dim0_stride[31:0] = row pitch
    g1[6] = 0;                            // stride[47:32] | dim1_stride[15:0]
    g1[7] = 0;                            // dim1_stride[47:16]
    i32x4 g2 = {0, 0, 0, 0};              // dims 2/3 unused (2-D tensor)
    i32x4 g3 = {0, 0, 0, 0};
    i32x8 g4 = {0, 0, 0, 0, 0, 0, 0, 0};  // extra group (6-arg toolchain form)
    __builtin_amdgcn_tensor_load_to_lds(g0, g1, g2, g3, g4, 0);
    __builtin_amdgcn_s_wait_tensorcnt(0);
  }
  __syncthreads();

  const int lane = tid & 31;
  const int wave = tid >> 5;      // wave32: 8 waves
  const int r    = lane & 15;
  const int hi   = lane >> 4;

  v8f acc[NTILES] = {};

#pragma unroll
  for (int tap = 0; tap < TAPS; ++tap) {
    const int kh = tap / 5, kw = tap % 5;
    // A fragment (16x32 f16): lane row M=r -> ox = 16*wave + r; K = channel.
    // Two aligned 16B LDS chunks at channel offsets 16*hi and 16*hi+8.
    const int x = wave * 16 + r + kw;
    const _Float16* ap = slab + ((kh * PADW + x) * CPAD + hi * 16);
    v8h alo = *reinterpret_cast<const v8h*>(ap);
    v8h ahi = *reinterpret_cast<const v8h*>(ap + 8);
    v16h a = __builtin_shufflevector(alo, ahi,
              0,1,2,3,4,5,6,7,8,9,10,11,12,13,14,15);

    const _Float16* bbase = bpack + ((long)(tap * NTILES) * 32 + lane) * 16;
    if (tap + 1 < TAPS)
      __builtin_prefetch(bpack + ((long)((tap + 1) * NTILES) * 32 + lane) * 16, 0, 3);

    // Preload all 7 B fragments for this tap so the WMMA burst overlaps loads.
    v16h bf[NTILES];
#pragma unroll
    for (int nt = 0; nt < NTILES; ++nt) {
      const _Float16* bp = bbase + nt * (32 * 16);
      v8h blo = *reinterpret_cast<const v8h*>(bp);
      v8h bhi = *reinterpret_cast<const v8h*>(bp + 8);
      bf[nt] = __builtin_shufflevector(blo, bhi,
                0,1,2,3,4,5,6,7,8,9,10,11,12,13,14,15);
    }
#pragma unroll
    for (int nt = 0; nt < NTILES; ++nt) {
      // reuse_a hint on all but the first of a burst of identical WMMAs that
      // share the same A registers (builtin requires literal-constant flags).
      if (nt == 0) {
        acc[nt] = __builtin_amdgcn_wmma_f32_16x16x32_f16(
            false, a, false, bf[nt], (short)0, acc[nt], false, false);
      } else {
        acc[nt] = __builtin_amdgcn_wmma_f32_16x16x32_f16(
            false, a, false, bf[nt], (short)0, acc[nt], true, false);
      }
    }
  }

  // fire(10) + per-WG any() into LDS flags. C/D layout: N = lane&15,
  // element j -> M row = j + 8*hi -> ox = 16*wave + j + 8*hi.
#pragma unroll
  for (int nt = 0; nt < NTILES; ++nt) {
    const int n = nt * 16 + r;
#pragma unroll
    for (int j = 0; j < 8; ++j) {
      const int ox = wave * 16 + hi * 8 + j;
      if (acc[nt][j] > 10.0f && ox < OH && n < 100) sflag[n] = 1;
    }
  }
  __syncthreads();
  if (tid < 100 && sflag[tid]) out[tid] = 1.0f;
}

// ---------------------------------------------------------------------------
extern "C" void kernel_launch(void* const* d_in, const int* in_sizes, int n_in,
                              void* d_out, int out_size, void* d_ws, size_t ws_size,
                              hipStream_t stream) {
  (void)in_sizes; (void)n_in; (void)out_size; (void)ws_size;
  const float* inp = (const float*)d_in[0];   // [32,2,256,256] f32 (binary)
  const float* w1  = (const float*)d_in[1];   // [30,2,5,5]
  const float* w2  = (const float*)d_in[2];   // [100,30,5,5]
  float* out = (float*)d_out;                 // [1,100] f32

  _Float16* pooled = (_Float16*)d_ws;                 // 36.5 MB channel-last f16
  _Float16* bpack  = pooled + POOLED_HALVES;          // 175 KB packed B fragments

  {
    long nvec = POOLED_HALVES / 8;
    int blocks = (int)((nvec + 255) / 256);
    prep_kernel<<<blocks, 256, 0, stream>>>(w2, pooled, bpack, out);
  }
  {
    long total = (long)T_DIM * POOL * POOL;
    int blocks = (int)((total + 255) / 256);
    conv1_fire_pool_kernel<<<blocks, 256, 0, stream>>>(inp, w1, pooled);
  }
  conv2_wmma_kernel<<<T_DIM * OH, 256, 0, stream>>>(pooled, bpack, out);
}
